// GAT_37649683316997
// MI455X (gfx1250) — matrix operand order repaired and checked by reading, not compiled, not run on hardware
//
#include <hip/hip_runtime.h>
#include <hip/hip_bf16.h>

#define NNODES 20000
#define NEDGES 640000
#define E2     (NEDGES + NNODES)   // edges + self loops
#define HIDC   64
#define HEADS  4
#define HC     (HEADS * HIDC)      // 256
#define OUTC   10
#define NGRAPH 128
#define NEG_SL 0.2f
#define BN_EPS 1e-5f

typedef __attribute__((ext_vector_type(16))) __bf16         v16bf;
typedef __attribute__((ext_vector_type(16))) unsigned short v16u;
typedef __attribute__((ext_vector_type(8)))  float          v8f;

union frag16 {            // 16 bf16 elements assembled from two 16B loads
  v16u  u;
  uint4 q[2];
};

static __device__ __forceinline__ unsigned short f32_bf16(float f) {
  unsigned int u = __float_as_uint(f);
  u += 0x7FFFu + ((u >> 16) & 1u);          // round-to-nearest-even
  return (unsigned short)(u >> 16);
}
static __device__ __forceinline__ float eluf(float v) {
  // branchless: exp(0)-1 == 0 exactly
  return fmaxf(v, 0.f) + __expf(fminf(v, 0.f)) - 1.f;
}
// monotone float <-> uint encoding for integer atomicMax
static __device__ __forceinline__ unsigned int f2ord(float f) {
  unsigned int i = __float_as_uint(f);
  return (i & 0x80000000u) ? ~i : (i | 0x80000000u);
}
static __device__ __forceinline__ float ord2f(unsigned int u) {
  unsigned int i = (u & 0x80000000u) ? (u & 0x7FFFFFFFu) : ~u;
  return __uint_as_float(i);
}

// ---------------- utility ----------------
__global__ void k_zero(float* p, int n) {
  int i = blockIdx.x * blockDim.x + threadIdx.x;
  if (i < n) p[i] = 0.f;
}

// fp32 -> bf16, element-wise (prep for layer-1 A operand)
__global__ void k_tobf16(const float* __restrict__ X, unsigned short* __restrict__ Y, int n) {
  int i = blockIdx.x * blockDim.x + threadIdx.x;
  if (i < n) Y[i] = f32_bf16(X[i]);
}

// W[K][C] fp32 row-major  ->  Wt[C][K] bf16 (column-major for contiguous K loads)
__global__ void k_wconv(const float* __restrict__ W, unsigned short* __restrict__ Wt,
                        int K, int C) {
  int i = blockIdx.x * blockDim.x + threadIdx.x;
  if (i >= K * C) return;
  int k = i / C, c = i % C;
  Wt[c * K + k] = f32_bf16(W[i]);
}

// ---------------- WMMA GEMM: Out[M][C] = A(bf16)[M][K] @ W ----------------
// one wave -> one 16x16 f32 tile; K compile-time so the loop fully unrolls
// into KSTEPS back-to-back v_wmma_f32_16x16x32_bf16 with pure b128 loads.
template <int K>
__global__ void k_gemm(const unsigned short* __restrict__ A,
                       const unsigned short* __restrict__ Wt,
                       float* __restrict__ Out, int M, int C) {
  const int lane  = threadIdx.x & 31;
  const int wave  = threadIdx.x >> 5;
  const int rbase = blockIdx.y * 128 + wave * 16;
  if (rbase >= M) return;
  const int tn  = blockIdx.x * 16;
  const int hi  = lane >> 4;                 // lane half selects K sub-range
  const int r   = rbase + (lane & 15);       // A row for this lane
  const int col = tn + (lane & 15);          // B/D column for this lane

  v8f acc = {};
  const unsigned short* arow = A  + (long)r   * K;
  const unsigned short* bcol = Wt + (long)col * K;

#pragma unroll
  for (int k0 = 0; k0 < K; k0 += 32) {
    // A fragment (16x32 bf16 layout): lanes 0-15 -> K {k0..+7, k0+16..+23},
    // lanes 16-31 shifted by 8. Two 16B loads.
    frag16 af, bf;
    af.q[0] = *reinterpret_cast<const uint4*>(arow + k0 + 8 * hi);
    af.q[1] = *reinterpret_cast<const uint4*>(arow + k0 + 16 + 8 * hi);
    // B fragment (32x16): same K order as A: {k0..+7, k0+16..+23}, hi half shifted by 8
    bf.q[0] = *reinterpret_cast<const uint4*>(bcol + k0 + 8 * hi);
    bf.q[1] = *reinterpret_cast<const uint4*>(bcol + k0 + 16 + 8 * hi);
    v16bf a = __builtin_bit_cast(v16bf, af.u);
    v16bf b = __builtin_bit_cast(v16bf, bf.u);
    acc = __builtin_amdgcn_wmma_f32_16x16x32_bf16(false, a, false, b,
                                                  (short)0, acc, false, false);
  }
#pragma unroll
  for (int v = 0; v < 8; ++v) {
    int row = rbase + v + 8 * hi;            // D layout: VGPR v -> M = v + 8*hi
    Out[(long)row * C + col] = acc[v];
  }
}

// ---------------- per-node attention logits: einsum('nhc,hc->nh') ----------------
__global__ void k_att(const float* __restrict__ H, const float* __restrict__ as_,
                      const float* __restrict__ ad_, float* __restrict__ atts,
                      float* __restrict__ attd) {
  int lane = threadIdx.x & 31;
  int wid  = (blockIdx.x * blockDim.x + threadIdx.x) >> 5;
  if (wid >= NNODES * HEADS) return;
  int n = wid >> 2, h = wid & 3;
  const float* hp = H + (long)n * HC + h * HIDC;
  float x0 = hp[lane], x1 = hp[lane + 32];
  float s = x0 * as_[h * HIDC + lane] + x1 * as_[h * HIDC + lane + 32];
  float d = x0 * ad_[h * HIDC + lane] + x1 * ad_[h * HIDC + lane + 32];
#pragma unroll
  for (int m = 16; m >= 1; m >>= 1) {
    s += __shfl_xor(s, m, 32);
    d += __shfl_xor(d, m, 32);
  }
  if (lane == 0) { atts[wid] = s; attd[wid] = d; }
}

// ---------------- edge pass 1: leaky-relu logits + segment max ----------------
__global__ void k_edge_max(const int* __restrict__ ei, const float* __restrict__ atts,
                           const float* __restrict__ attd, float* __restrict__ ebuf,
                           unsigned int* __restrict__ segm) {
  int idx = blockIdx.x * blockDim.x + threadIdx.x;
  if (idx >= E2 * HEADS) return;
  int e = idx >> 2, h = idx & 3;
  int s, d;
  if (e < NEDGES) { s = ei[e]; d = ei[NEDGES + e]; } else { s = d = e - NEDGES; }
  float v = atts[s * HEADS + h] + attd[d * HEADS + h];
  v = v > 0.f ? v : NEG_SL * v;
  ebuf[idx] = v;
  atomicMax(&segm[d * HEADS + h], f2ord(v));
}

// ---------------- edge pass 2: exp(e - max) + segment sum ----------------
__global__ void k_edge_exp(const int* __restrict__ ei, const unsigned int* __restrict__ segm,
                           float* __restrict__ ebuf, float* __restrict__ segd) {
  int idx = blockIdx.x * blockDim.x + threadIdx.x;
  if (idx >= E2 * HEADS) return;
  int e = idx >> 2, h = idx & 3;
  int d = (e < NEDGES) ? ei[NEDGES + e] : (e - NEDGES);
  float m = ord2f(segm[d * HEADS + h]);
  float w = __expf(ebuf[idx] - m);
  ebuf[idx] = w;
  atomicAdd(&segd[d * HEADS + h], w);
}

// ---------------- edge pass 3: scatter alpha * h[src] into h'[dst] ----------------
// one wave per edge, lane handles 8 consecutive channels (head = lane/8)
__global__ void k_scatter(const int* __restrict__ ei, const float* __restrict__ H,
                          const float* __restrict__ ebuf, const float* __restrict__ segd,
                          float* __restrict__ Outp) {
  int lane = threadIdx.x & 31;
  int e = (blockIdx.x * blockDim.x + threadIdx.x) >> 5;
  if (e >= E2) return;
  int s, d;
  if (e < NEDGES) { s = ei[e]; d = ei[NEDGES + e]; } else { s = d = e - NEDGES; }
  int h = lane >> 3;
  float alpha = ebuf[e * HEADS + h] / (segd[d * HEADS + h] + 1e-16f);
  const float4* src = reinterpret_cast<const float4*>(H + (long)s * HC + lane * 8);
  float4 v0 = src[0], v1 = src[1];
  float* dp = Outp + (long)d * HC + lane * 8;
  atomicAdd(dp + 0, v0.x * alpha); atomicAdd(dp + 1, v0.y * alpha);
  atomicAdd(dp + 2, v0.z * alpha); atomicAdd(dp + 3, v0.w * alpha);
  atomicAdd(dp + 4, v1.x * alpha); atomicAdd(dp + 5, v1.y * alpha);
  atomicAdd(dp + 6, v1.z * alpha); atomicAdd(dp + 7, v1.w * alpha);
}

// ---------------- BatchNorm (training-mode batch stats) ----------------
__global__ void k_bn_reduce(const float* __restrict__ X, float* __restrict__ sum,
                            float* __restrict__ sumsq) {
  int c = threadIdx.x;                       // 256 columns
  int r0 = blockIdx.x * 50;                  // 400 blocks * 50 rows = 20000
  float s = 0.f, s2 = 0.f;
  for (int r = r0; r < r0 + 50; ++r) {
    float v = X[(long)r * HC + c];
    s += v; s2 += v * v;
  }
  atomicAdd(&sum[c], s);
  atomicAdd(&sumsq[c], s2);
}
__global__ void k_bn_final(const float* sum, const float* sumsq, const float* g,
                           const float* be, float* scale, float* shift) {
  int c = threadIdx.x;
  float mu  = sum[c] * (1.f / NNODES);
  float var = sumsq[c] * (1.f / NNODES) - mu * mu;
  float sc  = g[c] * rsqrtf(var + BN_EPS);
  scale[c] = sc; shift[c] = be[c] - mu * sc;  // GAT bias cancels in BN exactly
}
// fused: y = BN(x) -> d_out ; bf16(elu(y)) -> next layer's A operand
__global__ void k_bn_apply(const float* __restrict__ X, const float* __restrict__ scale,
                           const float* __restrict__ shift, float* __restrict__ Y,
                           unsigned short* __restrict__ Abf) {
  int i = blockIdx.x * blockDim.x + threadIdx.x;
  if (i >= NNODES * HC) return;
  int c = i & 255;
  float y = X[i] * scale[c] + shift[c];
  Y[i]   = y;
  Abf[i] = f32_bf16(eluf(y));
}

// ---------------- layer-3 head mean + bias ----------------
__global__ void k_headmean(const float* __restrict__ X, const float* __restrict__ b3,
                           float* __restrict__ Y) {
  int i = blockIdx.x * blockDim.x + threadIdx.x;
  if (i >= NNODES * HIDC) return;
  int n = i >> 6, c = i & 63;
  const float* p = X + (long)n * HC;
  Y[i] = 0.25f * (p[c] + p[64 + c] + p[128 + c] + p[192 + c]) + b3[c];
}

// ---------------- global mean pool + final linear ----------------
__global__ void k_pool_acc(const float* __restrict__ Xs2, const int* __restrict__ batch,
                           float* __restrict__ psum) {
  int i = blockIdx.x * blockDim.x + threadIdx.x;
  if (i >= NNODES * HIDC) return;
  int n = i >> 6, c = i & 63;
  atomicAdd(&psum[batch[n] * HIDC + c], Xs2[i]);
}
__global__ void k_pool_cnt(const int* __restrict__ batch, float* __restrict__ cnt) {
  int n = blockIdx.x * blockDim.x + threadIdx.x;
  if (n >= NNODES) return;
  atomicAdd(&cnt[batch[n]], 1.f);
}
__global__ void k_pool_fin(const float* __restrict__ psum, const float* __restrict__ cnt,
                           float* __restrict__ pooled) {
  int i = blockIdx.x * blockDim.x + threadIdx.x;
  if (i >= NGRAPH * HIDC) return;
  pooled[i] = psum[i] / fmaxf(cnt[i >> 6], 1.f);
}
__global__ void k_linear(const float* __restrict__ pooled, const float* __restrict__ Wl,
                         const float* __restrict__ bl, float* __restrict__ out) {
  int i = blockIdx.x * blockDim.x + threadIdx.x;
  if (i >= NGRAPH * OUTC) return;
  int g = i / OUTC, o = i % OUTC;
  float s = bl[o];
#pragma unroll
  for (int c = 0; c < HIDC; ++c) s += pooled[g * HIDC + c] * Wl[c * OUTC + o];
  out[i] = s;
}

// ================= launcher =================
static inline int cdiv(long a, long b) { return (int)((a + b - 1) / b); }

extern "C" void kernel_launch(void* const* d_in, const int* in_sizes, int n_in,
                              void* d_out, int out_size, void* d_ws, size_t ws_size,
                              hipStream_t stream) {
  const float* x     = (const float*)d_in[0];
  const int*   ei    = (const int*)d_in[1];
  const int*   batch = (const int*)d_in[2];
  const float* W1  = (const float*)d_in[3];
  const float* as1 = (const float*)d_in[4];
  const float* ad1 = (const float*)d_in[5];
  const float* g1  = (const float*)d_in[7];
  const float* be1 = (const float*)d_in[8];
  const float* W2  = (const float*)d_in[9];
  const float* as2 = (const float*)d_in[10];
  const float* ad2 = (const float*)d_in[11];
  const float* g2  = (const float*)d_in[13];
  const float* be2 = (const float*)d_in[14];
  const float* W3  = (const float*)d_in[15];
  const float* as3 = (const float*)d_in[16];
  const float* ad3 = (const float*)d_in[17];
  const float* b3  = (const float*)d_in[18];
  const float* Wl  = (const float*)d_in[19];
  const float* bl  = (const float*)d_in[20];

  float* out = (float*)d_out;
  float* ws  = (float*)d_ws;

  // d_out layout (floats): out | xs0 | xs1 | xs2 | pooled
  float* o_out  = out;
  float* o_xs0  = out + 1280;
  float* o_xs1  = out + 5121280;
  float* o_xs2  = out + 10241280;
  float* o_pool = out + 11521280;

  // workspace layout (floats)
  float*        bufA  = ws;                       // 5,120,000  post-GEMM h
  float*        bufB  = ws + 5120000;             // 5,120,000  aggregated h'
  float*        atts  = ws + 10240000;            // 80,000
  float*        attd  = ws + 10320000;            // 80,000
  unsigned int* segm  = (unsigned int*)(ws + 10400000); // 80,000 (ordered uint)
  float*        segd  = ws + 10480000;            // 80,000
  float*        ebuf  = ws + 10560000;            // 2,640,000
  float*        bnsum = ws + 13200000;            // 256
  float*        bnsq  = ws + 13200256;            // 256
  float*        bnsc  = ws + 13200512;            // 256
  float*        bnsh  = ws + 13200768;            // 256
  float*        psum  = ws + 13201024;            // 8192
  float*        pcnt  = ws + 13209216;            // 128
  unsigned short* W1t = (unsigned short*)(ws + 13209344); // 128*256 bf16
  unsigned short* W2t = (unsigned short*)(ws + 13225728); // 256*256 bf16
  unsigned short* W3t = (unsigned short*)(ws + 13258496); // 256*256 bf16
  unsigned short* xbf = (unsigned short*)(ws + 13291264); // 20000*128 bf16
  unsigned short* abf = (unsigned short*)(ws + 14571264); // 20000*256 bf16

  const int TB  = 256;
  const int nEH = E2 * HEADS;

  // --- operand prep: weights (bf16 col-major) + layer-1 A (bf16 row-major) ---
  k_wconv<<<cdiv(128 * 256, TB), TB, 0, stream>>>(W1, W1t, 128, 256);
  k_wconv<<<cdiv(256 * 256, TB), TB, 0, stream>>>(W2, W2t, 256, 256);
  k_wconv<<<cdiv(256 * 256, TB), TB, 0, stream>>>(W3, W3t, 256, 256);
  k_tobf16<<<cdiv(NNODES * 128, TB), TB, 0, stream>>>(x, xbf, NNODES * 128);

  const unsigned short* Ain[3]  = { xbf, abf, abf };
  const unsigned short* Wts[3]  = { W1t, W2t, W3t };
  const float*          asl[3]  = { as1, as2, as3 };
  const float*          adl[3]  = { ad1, ad2, ad3 };

  for (int l = 0; l < 3; ++l) {
    // GEMM h = A @ W   (pure bf16 WMMA, fully unrolled K)
    dim3 gg(HC / 16, cdiv(NNODES, 128));
    if (l == 0)
      k_gemm<128><<<gg, TB, 0, stream>>>(Ain[l], Wts[l], bufA, NNODES, HC);
    else
      k_gemm<256><<<gg, TB, 0, stream>>>(Ain[l], Wts[l], bufA, NNODES, HC);

    // attention logits per (node, head)
    k_att<<<cdiv((long)NNODES * HEADS * 32, TB), TB, 0, stream>>>(bufA, asl[l], adl[l], atts, attd);

    // reset segment state + accumulator
    k_zero<<<cdiv(NNODES * HEADS, TB), TB, 0, stream>>>((float*)segm, NNODES * HEADS);
    k_zero<<<cdiv(NNODES * HEADS, TB), TB, 0, stream>>>(segd, NNODES * HEADS);
    k_zero<<<cdiv(NNODES * HC, TB), TB, 0, stream>>>(bufB, NNODES * HC);

    // segment softmax over dst + scatter-add
    k_edge_max<<<cdiv(nEH, TB), TB, 0, stream>>>(ei, atts, attd, ebuf, segm);
    k_edge_exp<<<cdiv(nEH, TB), TB, 0, stream>>>(ei, segm, ebuf, segd);
    k_scatter<<<cdiv((long)E2 * 32, TB), TB, 0, stream>>>(ei, bufA, ebuf, segd, bufB);

    if (l < 2) {
      float* xsout = (l == 0) ? o_xs0 : o_xs1;
      const float* g  = (l == 0) ? g1  : g2;
      const float* be = (l == 0) ? be1 : be2;
      k_zero<<<cdiv(512, TB), TB, 0, stream>>>(bnsum, 512);  // bnsum+bnsq contiguous
      k_bn_reduce<<<400, 256, 0, stream>>>(bufB, bnsum, bnsq);
      k_bn_final<<<1, 256, 0, stream>>>(bnsum, bnsq, g, be, bnsc, bnsh);
      // fused BN -> xs output + bf16(elu) A operand for next layer
      k_bn_apply<<<cdiv(NNODES * HC, TB), TB, 0, stream>>>(bufB, bnsc, bnsh, xsout, abf);
    } else {
      k_headmean<<<cdiv(NNODES * HIDC, TB), TB, 0, stream>>>(bufB, b3, o_xs2);
    }
  }

  // global mean pool + linear head
  k_zero<<<cdiv(NGRAPH * HIDC + NGRAPH, TB), TB, 0, stream>>>(psum, NGRAPH * HIDC + NGRAPH);
  k_pool_acc<<<cdiv(NNODES * HIDC, TB), TB, 0, stream>>>(o_xs2, batch, psum);
  k_pool_cnt<<<cdiv(NNODES, TB), TB, 0, stream>>>(batch, pcnt);
  k_pool_fin<<<cdiv(NGRAPH * HIDC, TB), TB, 0, stream>>>(psum, pcnt, o_pool);
  k_linear<<<cdiv(NGRAPH * OUTC, TB), TB, 0, stream>>>(o_pool, Wl, bl, o_out);
}
